// AttentionBlock3D_31834297598012
// MI455X (gfx1250) — matrix operand order repaired and checked by reading, not compiled, not run on hardware
//
#include <hip/hip_runtime.h>
#include <hip/hip_bf16.h>

// ---------------------------------------------------------------------------
// AttentionBlock3D for MI455X (gfx1250, wave32, WMMA f16->f32)
// B=2, C=512, N = T*H*W = 4096, NH=8, hd=64, GROUPS=32
// ---------------------------------------------------------------------------

#define BB   2
#define CC   512
#define NN   4096
#define NH   8
#define HD   64
#define NGRP 32

#define LOG2E 1.44269504088896340736f

typedef __attribute__((ext_vector_type(16))) _Float16 v16h;
typedef __attribute__((ext_vector_type(8)))  _Float16 v8h;
typedef __attribute__((ext_vector_type(8)))  float    v8f;

__device__ __forceinline__ v16h cat8(v8h lo, v8h hi) {
    return __builtin_shufflevector(lo, hi, 0,1,2,3,4,5,6,7,8,9,10,11,12,13,14,15);
}

// A-fragment (16x32 f16, MxK) from row-major src, leading dim ld (elements).
// Lane l: row m0+(l&15); halves 0..7 = K kb..kb+7, halves 8..15 = K kb+16..kb+23,
// kb = k0 + (l>=16 ? 8 : 0).  (ISA 7.12.2, 16-bit A 16x32)
__device__ __forceinline__ v16h load_a_rowmajor(const _Float16* __restrict__ src,
                                                int ld, int m0, int k0, int lane) {
    int m  = m0 + (lane & 15);
    int kb = k0 + ((lane >> 4) << 3);
    const _Float16* p = src + (size_t)m * ld + kb;
    v8h lo = *(const v8h*)(p);
    v8h hi = *(const v8h*)(p + 16);
    return cat8(lo, hi);
}

// B-fragment (32x16 f16, KxN) where B[k][n] = src[(n0+n)*ld + k0+k] (K contiguous).
// Lane l: col n0+(l&15); halves 0..7 = K kb..kb+7, halves 8..15 = K kb+16..kb+23, kb = (l>=16?8:0).
__device__ __forceinline__ v16h load_b_kcontig(const _Float16* __restrict__ src,
                                               int ld, int n0, int k0, int lane) {
    int n  = n0 + (lane & 15);
    int kb = k0 + ((lane >> 4) << 3);
    const _Float16* p = src + (size_t)n * ld + kb;
    v8h lo = *(const v8h*)(p);
    v8h hi = *(const v8h*)(p + 16);
    return cat8(lo, hi);
}

__device__ __forceinline__ v8f wmma_f16(v16h a, v16h b, v8f c) {
    return __builtin_amdgcn_wmma_f32_16x16x32_f16(false, a, false, b,
                                                  (short)0, c, false, false);
}

// ---------------------------------------------------------------------------
// Kernel 1: GroupNorm(32, C) -> hT[b][n][c] (f16, transposed for K-contiguous B)
// ---------------------------------------------------------------------------
__global__ __launch_bounds__(256)
void gn_kernel(const float* __restrict__ x, const float* __restrict__ gw,
               const float* __restrict__ gb, _Float16* __restrict__ hT) {
    int bg = blockIdx.x;            // 0..63
    int b  = bg >> 5;
    int g  = bg & 31;
    const float* xp = x + ((size_t)b * CC + (size_t)g * 16) * NN;   // 65536 floats
    __shared__ float s_sum[256], s_sq[256];
    float sum = 0.f, sq = 0.f;
    for (int i = threadIdx.x; i < 16 * NN; i += 256) {
        float t = xp[i];
        sum += t; sq += t * t;
    }
    s_sum[threadIdx.x] = sum;
    s_sq [threadIdx.x] = sq;
    __syncthreads();
    for (int s = 128; s > 0; s >>= 1) {
        if ((int)threadIdx.x < s) {
            s_sum[threadIdx.x] += s_sum[threadIdx.x + s];
            s_sq [threadIdx.x] += s_sq [threadIdx.x + s];
        }
        __syncthreads();
    }
    float mean = s_sum[0] * (1.0f / 65536.0f);
    float var  = s_sq[0]  * (1.0f / 65536.0f) - mean * mean;
    float rstd = rsqrtf(var + 1e-5f);
    for (int i = threadIdx.x; i < 16 * NN; i += 256) {
        int cl = i >> 12;
        int n  = i & (NN - 1);
        int c  = g * 16 + cl;
        float t = (xp[i] - mean) * rstd * gw[c] + gb[c];
        hT[((size_t)b * NN + n) * CC + c] = (_Float16)t;
    }
}

// ---------------------------------------------------------------------------
// Kernel 2: weights fp32 -> f16
// ---------------------------------------------------------------------------
__global__ __launch_bounds__(256)
void cvt_kernel(const float* __restrict__ qkvw, const float* __restrict__ outw,
                _Float16* __restrict__ wq, _Float16* __restrict__ wo) {
    int i = blockIdx.x * 256 + threadIdx.x;
    if (i < 3 * CC * CC) wq[i] = (_Float16)qkvw[i];
    if (i < CC * CC)     wo[i] = (_Float16)outw[i];
}

// ---------------------------------------------------------------------------
// Kernel 3: QKV GEMM  qkv[b][o][n] = qkv_w[o][:] . h[b][:][n] + bias
// Wave job: 16(o) x 64(n) tile, K-loop over C step 32.
// q pre-scaled by (1/sqrt(hd))*log2(e) (exp2 softmax domain).
// q,k stored [B][NH][N][HD]; v stored transposed [B][NH][HD][N].
// ---------------------------------------------------------------------------
__global__ __launch_bounds__(256)
void qkv_kernel(const _Float16* __restrict__ hT, const _Float16* __restrict__ wq,
                const float* __restrict__ bias,
                _Float16* __restrict__ qd, _Float16* __restrict__ kd,
                _Float16* __restrict__ vT) {
    int lane = threadIdx.x & 31;
    int job  = blockIdx.x * 8 + (threadIdx.x >> 5);   // 12288 jobs
    int b  = job / (96 * 64);
    int r2 = job % (96 * 64);
    int o0 = (r2 / 64) * 16;
    int n0 = (r2 % 64) * 64;
    const _Float16* hb = hT + (size_t)b * NN * CC;
    const v8f vz = {0.f,0.f,0.f,0.f,0.f,0.f,0.f,0.f};
    v8f acc[4] = {vz, vz, vz, vz};
    for (int k0 = 0; k0 < CC; k0 += 32) {
        v16h a = load_a_rowmajor(wq, CC, o0, k0, lane);
#pragma unroll
        for (int t = 0; t < 4; ++t) {
            v16h bf = load_b_kcontig(hb, CC, n0 + 16 * t, k0, lane);
            acc[t] = wmma_f16(a, bf, acc[t]);
        }
    }
    // wave-uniform decode: 16-row o-tile never crosses a j(512) or head(64) boundary
    int j    = o0 >> 9;                  // 0=q 1=k 2=v
    int ci0  = o0 & (CC - 1);
    int head = ci0 >> 6;
    int d0   = ci0 & (HD - 1);
    size_t bh = (size_t)b * NH + head;
    int half = lane >> 4, col = lane & 15;
    if (j < 2) {
        _Float16* dst = (j == 0) ? qd : kd;
        float scl     = (j == 0) ? (0.125f * LOG2E) : 1.0f;
#pragma unroll
        for (int t = 0; t < 4; ++t) {
            int n = n0 + 16 * t + col;
#pragma unroll
            for (int r = 0; r < 8; ++r) {
                int rr = half * 8 + r;
                float val = (acc[t][r] + bias[o0 + rr]) * scl;
                dst[(bh * NN + n) * HD + d0 + rr] = (_Float16)val;
            }
        }
    } else {
#pragma unroll
        for (int t = 0; t < 4; ++t) {
            int n = n0 + 16 * t + col;
#pragma unroll
            for (int r = 0; r < 8; ++r) {
                int rr = half * 8 + r;
                float val = acc[t][r] + bias[o0 + rr];
                vT[(bh * HD + d0 + rr) * NN + n] = (_Float16)val;   // transposed
            }
        }
    }
}

// ---------------------------------------------------------------------------
// Kernel 4: flash attention per (b, head). Wave owns 16 query rows, online
// softmax (exp2 domain) over key blocks of 64. P restaged via wave-private
// LDS to convert C-layout (v8f) -> A-layout (v16h) for the PV WMMAs.
// V read transposed -> all fragments are contiguous b128 loads.
// ---------------------------------------------------------------------------
__global__ __launch_bounds__(256)
void attn_kernel(const _Float16* __restrict__ qd, const _Float16* __restrict__ kd,
                 const _Float16* __restrict__ vT, _Float16* __restrict__ obuf) {
    __shared__ __align__(16) _Float16 pbuf[8 * 16 * 64];   // 8 waves x 16x64 P
    int lane = threadIdx.x & 31;
    int w    = threadIdx.x >> 5;
    _Float16* pw = pbuf + w * (16 * 64);
    int blk = blockIdx.x;                                   // 512
    int b  = blk / (NH * 32);
    int rr = blk % (NH * 32);
    int h  = rr / 32;
    int qb = rr % 32;
    int qrow0 = qb * 128 + w * 16;
    const _Float16* qp  = qd + (((size_t)b * NH + h) * NN) * HD;
    const _Float16* kp  = kd + (((size_t)b * NH + h) * NN) * HD;
    const _Float16* vpT = vT + (((size_t)b * NH + h) * HD) * NN;

    v16h aq0 = load_a_rowmajor(qp, HD, qrow0, 0,  lane);
    v16h aq1 = load_a_rowmajor(qp, HD, qrow0, 32, lane);

    const v8f vz = {0.f,0.f,0.f,0.f,0.f,0.f,0.f,0.f};
    v8f oacc[4] = {vz, vz, vz, vz};
    float mi[8], li[8];
#pragma unroll
    for (int r = 0; r < 8; ++r) { mi[r] = -1e30f; li[r] = 0.f; }

    int half = lane >> 4, col = lane & 15;

    for (int kb = 0; kb < NN; kb += 64) {
        // S tiles (exp2 domain; q pre-scaled): four 16x16 tiles over 64 keys
        v8f s[4];
#pragma unroll
        for (int t = 0; t < 4; ++t) {
            v8f c = vz;
            c = wmma_f16(aq0, load_b_kcontig(kp, HD, kb + 16 * t, 0,  lane), c);
            c = wmma_f16(aq1, load_b_kcontig(kp, HD, kb + 16 * t, 32, lane), c);
            s[t] = c;
        }
        // online softmax; row = 8*half + r, spread across a 16-lane half
#pragma unroll
        for (int r = 0; r < 8; ++r) {
            float mx = fmaxf(fmaxf(s[0][r], s[1][r]), fmaxf(s[2][r], s[3][r]));
            mx = fmaxf(mx, __shfl_xor(mx, 1, 16));
            mx = fmaxf(mx, __shfl_xor(mx, 2, 16));
            mx = fmaxf(mx, __shfl_xor(mx, 4, 16));
            mx = fmaxf(mx, __shfl_xor(mx, 8, 16));
            float mnew = fmaxf(mi[r], mx);
            float corr = __builtin_amdgcn_exp2f(mi[r] - mnew);
            float p0 = __builtin_amdgcn_exp2f(s[0][r] - mnew);
            float p1 = __builtin_amdgcn_exp2f(s[1][r] - mnew);
            float p2 = __builtin_amdgcn_exp2f(s[2][r] - mnew);
            float p3 = __builtin_amdgcn_exp2f(s[3][r] - mnew);
            float rs = (p0 + p1) + (p2 + p3);
            rs += __shfl_xor(rs, 1, 16);
            rs += __shfl_xor(rs, 2, 16);
            rs += __shfl_xor(rs, 4, 16);
            rs += __shfl_xor(rs, 8, 16);
            li[r] = li[r] * corr + rs;
            mi[r] = mnew;
            s[0][r] = p0; s[1][r] = p1; s[2][r] = p2; s[3][r] = p3;
            oacc[0][r] *= corr; oacc[1][r] *= corr;
            oacc[2][r] *= corr; oacc[3][r] *= corr;
        }
        // stage P (f16, row-major 16x64) in wave-private LDS
#pragma unroll
        for (int r = 0; r < 8; ++r) {
            int m = half * 8 + r;
#pragma unroll
            for (int t = 0; t < 4; ++t)
                pw[m * 64 + t * 16 + col] = (_Float16)s[t][r];
        }
        asm volatile("s_wait_dscnt 0x0" ::: "memory");
        // P as two A-fragments (cols 0..31 and 32..63), O += P @ V
        {
            int m = lane & 15, kk = (lane >> 4) << 3;
            const _Float16* pp = pw + m * 64 + kk;
            v16h pa0 = cat8(*(const v8h*)(pp),      *(const v8h*)(pp + 16));
#pragma unroll
            for (int jd = 0; jd < 4; ++jd) {
                v16h bv = load_b_kcontig(vpT, NN, jd * 16, kb, lane);
                oacc[jd] = wmma_f16(pa0, bv, oacc[jd]);
            }
            v16h pa1 = cat8(*(const v8h*)(pp + 32), *(const v8h*)(pp + 48));
#pragma unroll
            for (int jd = 0; jd < 4; ++jd) {
                v16h bv = load_b_kcontig(vpT, NN, jd * 16, kb + 32, lane);
                oacc[jd] = wmma_f16(pa1, bv, oacc[jd]);
            }
        }
    }
    // normalize and write O f16 [B][NH][N][HD]
#pragma unroll
    for (int r = 0; r < 8; ++r) {
        float inv = 1.0f / li[r];
        int m = qrow0 + half * 8 + r;
#pragma unroll
        for (int jd = 0; jd < 4; ++jd) {
            obuf[(((size_t)b * NH + h) * NN + m) * HD + jd * 16 + col] =
                (_Float16)(oacc[jd][r] * inv);
        }
    }
}

// ---------------------------------------------------------------------------
// Kernel 5: out projection + bias + residual (f32 output)
// ---------------------------------------------------------------------------
__global__ __launch_bounds__(256)
void outproj_kernel(const _Float16* __restrict__ obuf, const _Float16* __restrict__ wo,
                    const float* __restrict__ ob, const float* __restrict__ x,
                    float* __restrict__ out) {
    int lane = threadIdx.x & 31;
    int job  = blockIdx.x * 8 + (threadIdx.x >> 5);   // 4096 jobs
    int b  = job / (32 * 64);
    int r2 = job % (32 * 64);
    int o0 = (r2 / 64) * 16;
    int n0 = (r2 % 64) * 64;
    const v8f vz = {0.f,0.f,0.f,0.f,0.f,0.f,0.f,0.f};
    v8f acc[4] = {vz, vz, vz, vz};
    for (int k0 = 0; k0 < CC; k0 += 32) {
        v16h a = load_a_rowmajor(wo, CC, o0, k0, lane);
        int head  = k0 >> 6;
        int dbase = k0 & (HD - 1);          // 32-chunks never cross a head
        const _Float16* obp = obuf + (((size_t)b * NH + head) * NN) * HD + dbase;
#pragma unroll
        for (int t = 0; t < 4; ++t) {
            v16h bf = load_b_kcontig(obp, HD, n0 + 16 * t, 0, lane);
            acc[t] = wmma_f16(a, bf, acc[t]);
        }
    }
    int half = lane >> 4, col = lane & 15;
#pragma unroll
    for (int t = 0; t < 4; ++t) {
        int n = n0 + 16 * t + col;
#pragma unroll
        for (int r = 0; r < 8; ++r) {
            int o = o0 + half * 8 + r;
            size_t idx = ((size_t)b * CC + o) * NN + n;
            out[idx] = x[idx] + acc[t][r] + ob[o];
        }
    }
}

// ---------------------------------------------------------------------------
// Launch
// ---------------------------------------------------------------------------
extern "C" void kernel_launch(void* const* d_in, const int* in_sizes, int n_in,
                              void* d_out, int out_size, void* d_ws, size_t ws_size,
                              hipStream_t stream) {
    const float* x     = (const float*)d_in[0];
    const float* gn_w  = (const float*)d_in[1];
    const float* gn_b  = (const float*)d_in[2];
    const float* qkv_w = (const float*)d_in[3];
    const float* qkv_b = (const float*)d_in[4];
    const float* out_w = (const float*)d_in[5];
    const float* out_b = (const float*)d_in[6];
    float* out = (float*)d_out;

    // workspace layout (f16 elements): hT | wq | wo | q | k | vT ; O reuses hT
    _Float16* hT = (_Float16*)d_ws;                 // B*N*C      = 4194304
    _Float16* wq = hT + (size_t)BB * NN * CC;       // 3C*C       =  786432
    _Float16* wo = wq + (size_t)3 * CC * CC;        // C*C        =  262144
    _Float16* qd = wo + (size_t)CC * CC;            // B*NH*N*HD  = 4194304
    _Float16* kd = qd + (size_t)BB * NH * NN * HD;
    _Float16* vT = kd + (size_t)BB * NH * NN * HD;  // transposed [B][NH][HD][N]
    _Float16* ob = hT;                              // reuse hT for attention O

    gn_kernel     <<<BB * NGRP, 256, 0, stream>>>(x, gn_w, gn_b, hT);
    cvt_kernel    <<<(3 * CC * CC) / 256, 256, 0, stream>>>(qkv_w, out_w, wq, wo);
    qkv_kernel    <<<1536, 256, 0, stream>>>(hT, wq, qkv_b, qd, kd, vT);
    attn_kernel   <<<512, 256, 0, stream>>>(qd, kd, vT, ob);
    outproj_kernel<<<512, 256, 0, stream>>>(ob, wo, out_b, x, out);
}
